// SwinTransformerBlock_81295140979154
// MI455X (gfx1250) — matrix operand order repaired and checked by reading, not compiled, not run on hardware
//
#include <hip/hip_runtime.h>
#include <hip/hip_bf16.h>

typedef unsigned short u16;
typedef unsigned int   u32;
typedef __attribute__((ext_vector_type(16))) __bf16 v16bf;
typedef __attribute__((ext_vector_type(8)))  float  v8f;

union FragBF { v16bf v; u32 u[8]; };

// ---------- constants ----------
#define BATCH   32
#define CDIM    192
#define HW      112
#define POS     12544        // 112*112
#define NTOK    401408       // 32*112*112
#define NWIN    8192         // 32*16*16
#define TOK     49
#define HEADS   6
#define HD      32
#define HID     768
#define QKVC    576

__device__ __forceinline__ u16 f2bf(float f) {
    u32 u = __float_as_uint(f);
    u32 r = u + 0x7fffu + ((u >> 16) & 1u);
    return (u16)(r >> 16);
}
__device__ __forceinline__ float rsum16(float v) {
    v += __shfl_xor(v, 1, 16); v += __shfl_xor(v, 2, 16);
    v += __shfl_xor(v, 4, 16); v += __shfl_xor(v, 8, 16);
    return v;
}
__device__ __forceinline__ float rmax16(float v) {
    v = fmaxf(v, __shfl_xor(v, 1, 16)); v = fmaxf(v, __shfl_xor(v, 2, 16));
    v = fmaxf(v, __shfl_xor(v, 4, 16)); v = fmaxf(v, __shfl_xor(v, 8, 16));
    return v;
}

// ---------- K0: weight conversion f32 -> bf16 ----------
__global__ void k_cvt_weights(const float* wq, const float* wp, const float* w1, const float* w2,
                              u16* oq, u16* op, u16* o1, u16* o2) {
    int i = blockIdx.x * 256 + threadIdx.x;
    // sizes: 110592, 36864, 147456, 147456  (total 442368)
    if (i < 110592)       { oq[i] = f2bf(wq[i]); return; }
    i -= 110592;
    if (i < 36864)        { op[i] = f2bf(wp[i]); return; }
    i -= 36864;
    if (i < 147456)       { o1[i] = f2bf(w1[i]); return; }
    i -= 147456;
    if (i < 147456)       { o2[i] = f2bf(w2[i]); }
}

// ---------- K1: LN1 + roll(-3,-3) + window partition -> bf16 rows ----------
__global__ void k_ln1_shift_window(const float* __restrict__ x, const float* __restrict__ g1,
                                   const float* __restrict__ b1, u16* __restrict__ xw) {
    __shared__ float tile[CDIM * 17];            // [c][ww], stride 17 (bank-safe)
    const int tid = threadIdx.x;
    const int w0 = blockIdx.x * 16;
    const int h  = blockIdx.y;
    const int b  = blockIdx.z;
    // load 192x16 tile, coalesced along w
    #pragma unroll
    for (int it = 0; it < 12; ++it) {
        int idx = tid + it * 256;                // 0..3071
        int c = idx >> 4, ww = idx & 15;
        tile[c * 17 + ww] = x[(size_t)(b * CDIM + c) * POS + h * HW + w0 + ww];
    }
    __syncthreads();
    const int grp = tid >> 4;                    // position within tile
    const int lk  = tid & 15;                    // lane within group
    float xv[12]; float s = 0.f;
    #pragma unroll
    for (int i = 0; i < 12; ++i) { xv[i] = tile[(lk * 12 + i) * 17 + grp]; s += xv[i]; }
    s = rsum16(s);
    float mu = s * (1.0f / CDIM);
    float s2 = 0.f;
    #pragma unroll
    for (int i = 0; i < 12; ++i) { float d = xv[i] - mu; s2 += d * d; }
    s2 = rsum16(s2);
    float rs = rsqrtf(s2 * (1.0f / CDIM) + 1e-5f);
    // destination: shifted + windowed
    int w  = w0 + grp;
    int hs = (h >= 3) ? h - 3 : h + 109;
    int ws = (w >= 3) ? w - 3 : w + 109;
    int wr = hs / 7, wc = ws / 7;
    int row = ((b * 16 + wr) * 16 + wc) * TOK + (hs % 7) * 7 + (ws % 7);
    u16* dst = xw + (size_t)row * CDIM + lk * 12;
    #pragma unroll
    for (int i = 0; i < 12; ++i) {
        int c = lk * 12 + i;
        dst[i] = f2bf((xv[i] - mu) * rs * g1[c] + b1[c]);
    }
}

// ---------- WMMA GEMM core: 128x64 block tile, 8 waves of 32x32 ----------
__device__ __forceinline__ void gemm_core(const u16* __restrict__ A, const u16* __restrict__ B,
                                          int K, int N, int Mb, int Nb,
                                          u32* ldsA, u32* ldsB, v8f* acc /*[4]*/) {
    const int tid  = threadIdx.x;
    const int lane = tid & 31;
    const int wid  = tid >> 5;
    const int wm = wid >> 1, wn = wid & 1;
    const int g = lane >> 4, lr = lane & 15;
    #pragma unroll
    for (int q = 0; q < 4; ++q)
        #pragma unroll
        for (int e = 0; e < 8; ++e) acc[q][e] = 0.f;

    for (int kk = 0; kk < K; kk += 32) {
        __syncthreads();
        {   // stage A: 128x32 bf16, row stride 34 halves (17 u32)
            int row = tid >> 1;
            int col = (tid & 1) * 16;
            const u32* gp = (const u32*)(A + (size_t)(Mb + row) * K + kk + col);
            u32* dp = ldsA + row * 17 + (col >> 1);
            #pragma unroll
            for (int i = 0; i < 8; ++i) dp[i] = gp[i];
        }
        {   // stage B transposed: ldsB[n][k], stride 34 halves
            int kr = tid >> 3, nc = tid & 7;
            const u16* gp = B + (size_t)(kk + kr) * N + Nb + nc * 8;
            u16* dp = (u16*)ldsB;
            #pragma unroll
            for (int i = 0; i < 8; ++i) dp[(nc * 8 + i) * 34 + kr] = gp[i];
        }
        __syncthreads();
        FragBF af[2], bf[2];
        #pragma unroll
        for (int t = 0; t < 2; ++t) {
            int m = wm * 32 + t * 16 + lr;
            #pragma unroll
            for (int v = 0; v < 8; ++v) {
                int k = ((v < 4) ? 0 : 16) + g * 8 + (v & 3) * 2;
                af[t].u[v] = ldsA[m * 17 + (k >> 1)];
            }
            int n = wn * 32 + t * 16 + lr;
            #pragma unroll
            for (int v = 0; v < 8; ++v)
                bf[t].u[v] = ldsB[n * 17 + (((v < 4) ? 0 : 8) + g * 4 + (v & 3))];
        }
        #pragma unroll
        for (int i = 0; i < 2; ++i)
            #pragma unroll
            for (int j = 0; j < 2; ++j)
                acc[i * 2 + j] = __builtin_amdgcn_wmma_f32_16x16x32_bf16(
                    false, af[i].v, false, bf[j].v, (short)0, acc[i * 2 + j], false, false);
    }
}

// ---------- K2: QKV GEMM -> scattered Q/K/V ----------
__global__ void k_gemm_qkv(const u16* __restrict__ A, const u16* __restrict__ B,
                           const float* __restrict__ bias, u16* __restrict__ qkv) {
    __shared__ u32 ldsA[128 * 17];
    __shared__ u32 ldsB[64 * 17];
    v8f acc[4];
    const int Mb = blockIdx.x * 128, Nb = blockIdx.y * 64;
    gemm_core(A, B, CDIM, QKVC, Mb, Nb, ldsA, ldsB, acc);
    const int lane = threadIdx.x & 31, wid = threadIdx.x >> 5;
    const int wm = wid >> 1, wn = wid & 1, g = lane >> 4, lr = lane & 15;
    #pragma unroll
    for (int i = 0; i < 2; ++i)
        #pragma unroll
        for (int j = 0; j < 2; ++j)
            #pragma unroll
            for (int r = 0; r < 8; ++r) {
                int R  = Mb + wm * 32 + i * 16 + r + 8 * g;
                int Cc = Nb + wn * 32 + j * 16 + lr;
                float v = acc[i * 2 + j][r] + bias[Cc];
                int win = R / TOK, t = R - win * TOK;
                int sel = Cc / CDIM; int rem = Cc - sel * CDIM;
                int head = rem >> 5, d = rem & 31;
                size_t off = (((size_t)sel * NWIN * HEADS + (size_t)(win * HEADS + head)) * TOK + t) * HD + d;
                qkv[off] = f2bf(v);
            }
}

// ---------- K3: windowed attention (one WG per window*head, 4 waves) ----------
__global__ void k_attention(const u16* __restrict__ qkv, u16* __restrict__ attn_out) {
    __shared__ u32 Ks_u[64 * 17];      // K:  [s][d]  stride 34 halves
    __shared__ u32 Vt_u[32 * 33];      // Vt: [d][s]  stride 66 halves
    __shared__ u32 Ps_u[64 * 33];      // P:  [t][s]  stride 66 halves
    __shared__ int reg[64];
    const int head = blockIdx.x, win = blockIdx.y;
    const int tid = threadIdx.x, lane = tid & 31, wid = tid >> 5;
    const int g = lane >> 4, lr = lane & 15;
    const int m0 = wid * 16;
    const size_t base = (size_t)(win * HEADS + head) * TOK * HD;
    const u16* Qg = qkv + base;
    const u16* Kg = qkv + (size_t)NWIN * HEADS * TOK * HD + base;
    const u16* Vg = qkv + 2 * (size_t)NWIN * HEADS * TOK * HD + base;

    // stage K (zero-padded) and V transposed
    {
        int s = tid >> 1, half = tid & 1;
        u32* dk = Ks_u + s * 17 + half * 8;
        u16* vt = (u16*)Vt_u;
        if (s < TOK) {
            const u32* gk = (const u32*)(Kg + s * HD + half * 16);
            #pragma unroll
            for (int i = 0; i < 8; ++i) dk[i] = gk[i];
            #pragma unroll
            for (int i = 0; i < 16; ++i) { int d = half * 16 + i; vt[d * 66 + s] = Vg[s * HD + d]; }
        } else {
            #pragma unroll
            for (int i = 0; i < 8; ++i) dk[i] = 0u;
            #pragma unroll
            for (int i = 0; i < 16; ++i) { int d = half * 16 + i; vt[d * 66 + s] = 0; }
        }
    }
    if (tid < 64) {
        int t = tid;
        if (t < TOK) {
            int wr = (win >> 4) & 15, wc = win & 15;
            int hh = (wr * 7 + t / 7 + 3) % HW;
            int ww = (wc * 7 + t % 7 + 3) % HW;
            reg[t] = (hh / 7) * 16 + (ww / 7);
        } else reg[t] = -1;
    }
    // Q fragment straight from global (A-layout; rows >= 49 are zero)
    FragBF qf;
    {
        int t = m0 + lr;
        #pragma unroll
        for (int v = 0; v < 8; ++v) {
            int k = ((v < 4) ? 0 : 16) + g * 8 + (v & 3) * 2;
            qf.u[v] = (t < TOK) ? *(const u32*)(Qg + t * HD + k) : 0u;
        }
    }
    __syncthreads();

    // scores: 4 tiles of 16x16, K=32 (one WMMA each)
    v8f sacc[4];
    #pragma unroll
    for (int nt = 0; nt < 4; ++nt) {
        FragBF kf;
        int n = nt * 16 + lr;
        #pragma unroll
        for (int v = 0; v < 8; ++v) kf.u[v] = Ks_u[n * 17 + ((v < 4) ? 0 : 8) + g * 4 + (v & 3)];
        v8f z;
        #pragma unroll
        for (int e = 0; e < 8; ++e) z[e] = 0.f;
        sacc[nt] = __builtin_amdgcn_wmma_f32_16x16x32_bf16(false, qf.v, false, kf.v, (short)0, z, false, false);
    }
    // mask + softmax (rows live across 16 lanes: half-wave shuffles)
    const float scale = 0.17677669529663687f;   // 1/sqrt(32)
    int rowreg[8];
    #pragma unroll
    for (int r = 0; r < 8; ++r) { int m = m0 + r + 8 * g; rowreg[r] = reg[m < TOK ? m : 0]; }
    float val[4][8];
    #pragma unroll
    for (int nt = 0; nt < 4; ++nt) {
        int cr = reg[nt * 16 + lr];
        #pragma unroll
        for (int r = 0; r < 8; ++r)
            val[nt][r] = (cr == rowreg[r]) ? sacc[nt][r] * scale : -1e30f;
    }
    u16* ps = (u16*)Ps_u;
    #pragma unroll
    for (int r = 0; r < 8; ++r) {
        float mx = fmaxf(fmaxf(val[0][r], val[1][r]), fmaxf(val[2][r], val[3][r]));
        mx = rmax16(mx);
        float sm = 0.f;
        #pragma unroll
        for (int nt = 0; nt < 4; ++nt) { val[nt][r] = __expf(val[nt][r] - mx); sm += val[nt][r]; }
        sm = rsum16(sm);
        float inv = 1.0f / sm;
        int row = m0 + r + 8 * g;
        #pragma unroll
        for (int nt = 0; nt < 4; ++nt)
            ps[row * 66 + nt * 16 + lr] = f2bf(val[nt][r] * inv);
    }
    // out = P @ V : A from Ps (this wave's own rows), B from Vt
    v8f oacc[2];
    #pragma unroll
    for (int dt = 0; dt < 2; ++dt) {
        #pragma unroll
        for (int e = 0; e < 8; ++e) oacc[dt][e] = 0.f;
        #pragma unroll
        for (int kc = 0; kc < 2; ++kc) {
            FragBF pf, vf;
            #pragma unroll
            for (int v = 0; v < 8; ++v) {
                int k = kc * 32 + ((v < 4) ? 0 : 16) + g * 8 + (v & 3) * 2;
                pf.u[v] = Ps_u[(m0 + lr) * 33 + (k >> 1)];
            }
            int n = dt * 16 + lr;
            #pragma unroll
            for (int v = 0; v < 8; ++v)
                vf.u[v] = Vt_u[n * 33 + kc * 16 + ((v < 4) ? 0 : 8) + g * 4 + (v & 3)];
            oacc[dt] = __builtin_amdgcn_wmma_f32_16x16x32_bf16(false, pf.v, false, vf.v, (short)0, oacc[dt], false, false);
        }
    }
    #pragma unroll
    for (int dt = 0; dt < 2; ++dt)
        #pragma unroll
        for (int r = 0; r < 8; ++r) {
            int t = m0 + r + 8 * g;
            if (t < TOK) {
                int d = dt * 16 + lr;
                attn_out[(size_t)(win * TOK + t) * CDIM + head * HD + d] = f2bf(oacc[dt][r]);
            }
        }
}

// ---------- K4: proj GEMM + window reverse + unshift + residual ----------
__global__ void k_gemm_proj(const u16* __restrict__ A, const u16* __restrict__ B,
                            const float* __restrict__ bias, const float* __restrict__ x,
                            float* __restrict__ x_attn) {
    __shared__ u32 ldsA[128 * 17];
    __shared__ u32 ldsB[64 * 17];
    v8f acc[4];
    const int Mb = blockIdx.x * 128, Nb = blockIdx.y * 64;
    gemm_core(A, B, CDIM, CDIM, Mb, Nb, ldsA, ldsB, acc);
    const int lane = threadIdx.x & 31, wid = threadIdx.x >> 5;
    const int wm = wid >> 1, wn = wid & 1, g = lane >> 4, lr = lane & 15;
    #pragma unroll
    for (int i = 0; i < 2; ++i)
        #pragma unroll
        for (int j = 0; j < 2; ++j)
            #pragma unroll
            for (int r = 0; r < 8; ++r) {
                int R  = Mb + wm * 32 + i * 16 + r + 8 * g;
                int Cc = Nb + wn * 32 + j * 16 + lr;
                int win = R / TOK, t = R - win * TOK;
                int b = win >> 8, wr = (win >> 4) & 15, wc = win & 15;
                int h = wr * 7 + t / 7 + 3;  if (h >= HW) h -= HW;
                int w = wc * 7 + t % 7 + 3;  if (w >= HW) w -= HW;
                float v = acc[i * 2 + j][r] + bias[Cc]
                        + x[(size_t)(b * CDIM + Cc) * POS + h * HW + w];
                x_attn[(size_t)(b * POS + h * HW + w) * CDIM + Cc] = v;
            }
}

// ---------- K5: LN2 ----------
__global__ void k_ln2(const float* __restrict__ x_attn, const float* __restrict__ g2,
                      const float* __restrict__ b2, u16* __restrict__ h_in) {
    const int tid = threadIdx.x;
    const int grp = tid >> 4, lk = tid & 15;
    const int p = blockIdx.x * 16 + grp;
    const float* src = x_attn + (size_t)p * CDIM + lk * 12;
    float xv[12]; float s = 0.f;
    #pragma unroll
    for (int i = 0; i < 12; ++i) { xv[i] = src[i]; s += xv[i]; }
    s = rsum16(s);
    float mu = s * (1.0f / CDIM);
    float s2 = 0.f;
    #pragma unroll
    for (int i = 0; i < 12; ++i) { float d = xv[i] - mu; s2 += d * d; }
    s2 = rsum16(s2);
    float rs = rsqrtf(s2 * (1.0f / CDIM) + 1e-5f);
    u16* dst = h_in + (size_t)p * CDIM + lk * 12;
    #pragma unroll
    for (int i = 0; i < 12; ++i) {
        int c = lk * 12 + i;
        dst[i] = f2bf((xv[i] - mu) * rs * g2[c] + b2[c]);
    }
}

// ---------- K6: FC1 GEMM + exact GELU ----------
__global__ void k_gemm_fc1(const u16* __restrict__ A, const u16* __restrict__ B,
                           const float* __restrict__ bias, u16* __restrict__ h1) {
    __shared__ u32 ldsA[128 * 17];
    __shared__ u32 ldsB[64 * 17];
    v8f acc[4];
    const int Mb = blockIdx.x * 128, Nb = blockIdx.y * 64;
    gemm_core(A, B, CDIM, HID, Mb, Nb, ldsA, ldsB, acc);
    const int lane = threadIdx.x & 31, wid = threadIdx.x >> 5;
    const int wm = wid >> 1, wn = wid & 1, g = lane >> 4, lr = lane & 15;
    #pragma unroll
    for (int i = 0; i < 2; ++i)
        #pragma unroll
        for (int j = 0; j < 2; ++j)
            #pragma unroll
            for (int r = 0; r < 8; ++r) {
                int R  = Mb + wm * 32 + i * 16 + r + 8 * g;
                int Cc = Nb + wn * 32 + j * 16 + lr;
                float v = acc[i * 2 + j][r] + bias[Cc];
                v = 0.5f * v * (1.0f + erff(v * 0.70710678118654752f));
                h1[(size_t)R * HID + Cc] = f2bf(v);
            }
}

// ---------- K7: FC2 GEMM + residual + transpose to (B,C,H,W) ----------
__global__ void k_gemm_fc2(const u16* __restrict__ A, const u16* __restrict__ B,
                           const float* __restrict__ bias, const float* __restrict__ x_attn,
                           float* __restrict__ out) {
    __shared__ u32 ldsA[128 * 17];
    __shared__ u32 ldsB[64 * 17];
    v8f acc[4];
    const int Mb = blockIdx.x * 128, Nb = blockIdx.y * 64;
    gemm_core(A, B, HID, CDIM, Mb, Nb, ldsA, ldsB, acc);
    const int lane = threadIdx.x & 31, wid = threadIdx.x >> 5;
    const int wm = wid >> 1, wn = wid & 1, g = lane >> 4, lr = lane & 15;
    #pragma unroll
    for (int i = 0; i < 2; ++i)
        #pragma unroll
        for (int j = 0; j < 2; ++j)
            #pragma unroll
            for (int r = 0; r < 8; ++r) {
                int R  = Mb + wm * 32 + i * 16 + r + 8 * g;
                int Cc = Nb + wn * 32 + j * 16 + lr;
                float v = acc[i * 2 + j][r] + bias[Cc] + x_attn[(size_t)R * CDIM + Cc];
                int b = R / POS, p = R - b * POS;
                out[(size_t)(b * CDIM + Cc) * POS + p] = v;
            }
}

// ---------- launch ----------
extern "C" void kernel_launch(void* const* d_in, const int* in_sizes, int n_in,
                              void* d_out, int out_size, void* d_ws, size_t ws_size,
                              hipStream_t stream) {
    const float* x    = (const float*)d_in[0];
    const float* g1   = (const float*)d_in[1];
    const float* b1   = (const float*)d_in[2];
    const float* wqkv = (const float*)d_in[3];
    const float* bqkv = (const float*)d_in[4];
    const float* wpr  = (const float*)d_in[5];
    const float* bpr  = (const float*)d_in[6];
    const float* g2   = (const float*)d_in[7];
    const float* b2   = (const float*)d_in[8];
    const float* wf1  = (const float*)d_in[9];
    const float* bf1  = (const float*)d_in[10];
    const float* wf2  = (const float*)d_in[11];
    const float* bf2  = (const float*)d_in[12];
    float* out = (float*)d_out;

    // workspace layout (bytes, all 256-aligned); total = 1,079,869,440 B
    char* ws = (char*)d_ws;
    u16* wq_bf  = (u16*)(ws + 0);                 // 221184 B
    u16* wp_bf  = (u16*)(ws + 221184);            // 73728 B
    u16* w1_bf  = (u16*)(ws + 294912);            // 294912 B
    u16* w2_bf  = (u16*)(ws + 589824);            // 294912 B  -> ends 884736
    u16* xw     = (u16*)(ws + 884736);            // 154,140,672 B (reused as h_in)
    u16* h_in   = xw;
    float* x_attn = (float*)(ws + 155025408);     // 308,281,344 B
    char* regC  = ws + 463306752;                 // 616,562,688 B shared region
    u16* qkv      = (u16*)(regC);                 // Q|K|V, 3 x 154,140,672 B
    u16* attn_out = (u16*)(regC + 462422016);     // 154,140,672 B
    u16* h1       = (u16*)(regC);                 // 616,562,688 B (after QKV dead)

    // K0: weight conversion (442368 elements)
    k_cvt_weights<<<1728, 256, 0, stream>>>(wqkv, wpr, wf1, wf2, wq_bf, wp_bf, w1_bf, w2_bf);
    // K1: LN1 + shift + window partition
    k_ln1_shift_window<<<dim3(7, HW, BATCH), 256, 0, stream>>>(x, g1, b1, xw);
    // K2: QKV GEMM (401408 x 576, K=192)
    k_gemm_qkv<<<dim3(NTOK / 128, QKVC / 64), 256, 0, stream>>>(xw, wq_bf, bqkv, qkv);
    // K3: attention per (head, window)
    k_attention<<<dim3(HEADS, NWIN), 128, 0, stream>>>(qkv, attn_out);
    // K4: proj GEMM + reverse + residual
    k_gemm_proj<<<dim3(NTOK / 128, CDIM / 64), 256, 0, stream>>>(attn_out, wp_bf, bpr, x, x_attn);
    // K5: LN2
    k_ln2<<<NTOK / 16, 256, 0, stream>>>(x_attn, g2, b2, h_in);
    // K6: FC1 + GELU (401408 x 768, K=192)
    k_gemm_fc1<<<dim3(NTOK / 128, HID / 64), 256, 0, stream>>>(h_in, w1_bf, bf1, h1);
    // K7: FC2 + residual + transpose (401408 x 192, K=768)
    k_gemm_fc2<<<dim3(NTOK / 128, CDIM / 64), 256, 0, stream>>>(h1, w2_bf, bf2, x_attn, out);
}
